// LSTMFeatureExtractor_36472862277716
// MI455X (gfx1250) — matrix operand order repaired and checked by reading, not compiled, not run on hardware
//
#include <hip/hip_runtime.h>
#include <cstdint>
#include <cstddef>

// ---------------------------------------------------------------------------
// Types for CDNA5 WMMA (wave32, v_wmma_f32_16x16x32_bf16)
// ---------------------------------------------------------------------------
typedef __bf16 bf16_t;
typedef __attribute__((ext_vector_type(16))) __bf16 v16bf;
typedef __attribute__((ext_vector_type(8)))  float  v8f;

struct alignas(16) U128 { uint32_t w[4]; };
union Frag { v16bf v; U128 q[2]; };

enum { B_ = 256, T_ = 512, IN_ = 64, H_ = 512, G4_ = 2048, H2_ = 256, NC_ = 64 };

__device__ inline bf16_t f2bf(float f) {
  union { float f; uint32_t u; } v; v.f = f;
  uint32_t r = v.u + 0x7FFFu + ((v.u >> 16) & 1u);   // round-to-nearest-even
  uint16_t hs = (uint16_t)(r >> 16);
  bf16_t o; __builtin_memcpy(&o, &hs, 2); return o;
}

__device__ inline v8f vzero8() {
  v8f z = {0.f, 0.f, 0.f, 0.f, 0.f, 0.f, 0.f, 0.f};
  return z;
}

__device__ inline float sigmoidf_(float x) { return 1.f / (1.f + __expf(-x)); }

// ---------------------------------------------------------------------------
// Fused LSTM step: gates = xin_t @ w_ih^T + hprev @ w_hh^T + bias, then
// c = sigm(f)*c + sigm(i)*tanh(g); h = sigm(o)*tanh(c).
// Grid: (16 M-blocks, 4 J-blocks of 128 gate-cols). Block: 256 thr = 8 waves.
// Wave w: gate (w&3), 64-col half (w>>2), 4 fragments of 16x16.
// ---------------------------------------------------------------------------
__global__ __launch_bounds__(256) void lstm_step_kernel(
    const bf16_t* __restrict__ xin, long long xin_stride,
    const bf16_t* __restrict__ w_ih, int Kin,
    const bf16_t* __restrict__ hprev, long long h_stride,
    const bf16_t* __restrict__ w_hh,
    const float*  __restrict__ bias,
    float*        __restrict__ cstate,
    bf16_t*       __restrict__ hout, long long hout_stride)
{
  __shared__ float lds_g[4][16][128];   // 32 KB of the 320 KB WGP LDS

  const int tid  = threadIdx.x;
  const int wave = tid >> 5;
  const int lane = tid & 31;
  const int half = lane >> 4;           // K sub-range selector (wave32 halves)
  const int lid  = lane & 15;           // M (A/C) or N (B/C) index
  const int gi   = wave & 3;            // gate: 0=i 1=f 2=g 3=o
  const int jh   = wave >> 2;           // which 64-col half of the 128-col block
  const int mblock = blockIdx.x;        // 0..15
  const int jblock = blockIdx.y;        // 0..3
  const int rowbase = mblock * 16;
  const int hs8  = half * 8;
  const int hs16 = half * 16;

  int nbase[4];
#pragma unroll
  for (int f = 0; f < 4; ++f)
    nbase[f] = gi * H_ + jblock * 128 + jh * 64 + f * 16;

  v8f acc[4];
#pragma unroll
  for (int f = 0; f < 4; ++f) acc[f] = vzero8();

  // ---- x-part: gates += x_t @ w_ih^T  (Kin = 64 or 512) ----
  {
    const bf16_t* arow = xin + (long long)(rowbase + lid) * xin_stride;
    const bf16_t* wrow[4];
#pragma unroll
    for (int f = 0; f < 4; ++f)
      wrow[f] = w_ih + (size_t)(nbase[f] + lid) * (size_t)Kin;

    for (int k0 = 0; k0 < Kin; k0 += 32) {
      Frag a;
      a.q[0] = *(const U128*)(arow + k0 + hs8);
      a.q[1] = *(const U128*)(arow + k0 + 16 + hs8);
#pragma unroll
      for (int f = 0; f < 4; ++f) {
        Frag b;
        const bf16_t* bp = wrow[f] + k0 + hs8;
        b.q[0] = *(const U128*)(bp);
        b.q[1] = *(const U128*)(bp + 16);
        acc[f] = __builtin_amdgcn_wmma_f32_16x16x32_bf16(
            false, a.v, false, b.v, (short)0, acc[f], false, false);
      }
    }
  }

  // ---- h-part: gates += h_{t-1} @ w_hh^T  (K = 512); skipped at t==0 ----
  if (hprev != nullptr) {
    const bf16_t* arow = hprev + (long long)(rowbase + lid) * h_stride;
    const bf16_t* wrow[4];
#pragma unroll
    for (int f = 0; f < 4; ++f)
      wrow[f] = w_hh + (size_t)(nbase[f] + lid) * (size_t)H_;

    for (int k0 = 0; k0 < H_; k0 += 32) {
      Frag a;
      a.q[0] = *(const U128*)(arow + k0 + hs8);
      a.q[1] = *(const U128*)(arow + k0 + 16 + hs8);
#pragma unroll
      for (int f = 0; f < 4; ++f) {
        Frag b;
        const bf16_t* bp = wrow[f] + k0 + hs8;
        b.q[0] = *(const U128*)(bp);
        b.q[1] = *(const U128*)(bp + 16);
        acc[f] = __builtin_amdgcn_wmma_f32_16x16x32_bf16(
            false, a.v, false, b.v, (short)0, acc[f], false, false);
      }
    }
  }

  // ---- bias add + park gates in LDS (C layout: VGPR v -> M = v + half*8) ----
#pragma unroll
  for (int f = 0; f < 4; ++f) {
    const int jcol = jh * 64 + f * 16 + lid;     // local gate column 0..127
    const float bv = bias[nbase[f] + lid];
#pragma unroll
    for (int v = 0; v < 8; ++v)
      lds_g[gi][v + hs8][jcol] = acc[f][v] + bv;
  }
  __syncthreads();

  // ---- fused elementwise LSTM cell update: 16 rows x 128 cols per block ----
  for (int e = tid; e < 16 * 128; e += 256) {
    const int m = e >> 7;
    const int j = e & 127;
    float iv = sigmoidf_(lds_g[0][m][j]);
    float fv = sigmoidf_(lds_g[1][m][j]);
    float gv = tanhf(lds_g[2][m][j]);
    float ov = sigmoidf_(lds_g[3][m][j]);
    const int row = rowbase + m;
    const int jg  = jblock * 128 + j;
    const long long ci = (long long)row * H_ + jg;
    float c = fv * cstate[ci] + iv * gv;
    cstate[ci] = c;
    float hval = ov * tanhf(c);
    hout[(long long)row * hout_stride + jg] = f2bf(hval);
  }
}

// ---------------------------------------------------------------------------
// Generic head GEMM: out = act(A(bf16, MxK) @ W^T(bf16, NxK) + bias)
// One 16x16 output tile per wave.
// ---------------------------------------------------------------------------
__global__ __launch_bounds__(256) void head_gemm_kernel(
    const bf16_t* __restrict__ A, int lda,
    const bf16_t* __restrict__ W, const float* __restrict__ bias,
    int Ntiles, int K,
    float* __restrict__ outF, bf16_t* __restrict__ outB, int ldo,
    int relu, int total_tiles)
{
  const int wave = threadIdx.x >> 5;
  const int lane = threadIdx.x & 31;
  const int half = lane >> 4;
  const int lid  = lane & 15;
  const int tile = blockIdx.x * 8 + wave;     // wave-uniform
  if (tile >= total_tiles) return;
  const int mt = tile / Ntiles;
  const int nt = tile % Ntiles;

  v8f acc = vzero8();
  const bf16_t* arow = A + (size_t)(mt * 16 + lid) * (size_t)lda;
  const bf16_t* brow = W + (size_t)(nt * 16 + lid) * (size_t)K;
  const int hs8 = half * 8, hs16 = half * 16;

  for (int k0 = 0; k0 < K; k0 += 32) {
    Frag a, b;
    a.q[0] = *(const U128*)(arow + k0 + hs8);
    a.q[1] = *(const U128*)(arow + k0 + 16 + hs8);
    b.q[0] = *(const U128*)(brow + k0 + hs8);
    b.q[1] = *(const U128*)(brow + k0 + 16 + hs8);
    acc = __builtin_amdgcn_wmma_f32_16x16x32_bf16(
        false, a.v, false, b.v, (short)0, acc, false, false);
  }

  const float bv = bias[nt * 16 + lid];
  const int n = nt * 16 + lid;
#pragma unroll
  for (int v = 0; v < 8; ++v) {
    float val = acc[v] + bv;
    if (relu) val = fmaxf(val, 0.f);
    const int m = mt * 16 + v + half * 8;
    if (outF) outF[(size_t)m * ldo + n] = val;
    else      outB[(size_t)m * ldo + n] = f2bf(val);
  }
}

// ---------------------------------------------------------------------------
// Utility kernels (grid-stride, deterministic)
// ---------------------------------------------------------------------------
__global__ void f2bf_kernel(const float* __restrict__ s, bf16_t* __restrict__ d,
                            long long n) {
  long long i  = (long long)blockIdx.x * blockDim.x + threadIdx.x;
  long long st = (long long)gridDim.x * blockDim.x;
  for (; i < n; i += st) d[i] = f2bf(s[i]);
}

__global__ void addvec_kernel(const float* __restrict__ a,
                              const float* __restrict__ b,
                              float* __restrict__ d, long long n) {
  long long i  = (long long)blockIdx.x * blockDim.x + threadIdx.x;
  long long st = (long long)gridDim.x * blockDim.x;
  for (; i < n; i += st) d[i] = a[i] + b[i];
}

__global__ void zerof_kernel(float* __restrict__ d, long long n) {
  long long i  = (long long)blockIdx.x * blockDim.x + threadIdx.x;
  long long st = (long long)gridDim.x * blockDim.x;
  for (; i < n; i += st) d[i] = 0.f;
}

// ---------------------------------------------------------------------------
// Host-side orchestration
// ---------------------------------------------------------------------------
extern "C" void kernel_launch(void* const* d_in, const int* in_sizes, int n_in,
                              void* d_out, int out_size, void* d_ws, size_t ws_size,
                              hipStream_t stream) {
  (void)in_sizes; (void)n_in; (void)out_size; (void)ws_size;
  const float* x     = (const float*)d_in[0];
  const float* w_ih0 = (const float*)d_in[1];
  const float* w_hh0 = (const float*)d_in[2];
  const float* b_ih0 = (const float*)d_in[3];
  const float* b_hh0 = (const float*)d_in[4];
  const float* w_ih1 = (const float*)d_in[5];
  const float* w_hh1 = (const float*)d_in[6];
  const float* b_ih1 = (const float*)d_in[7];
  const float* b_hh1 = (const float*)d_in[8];
  const float* w1    = (const float*)d_in[9];
  const float* b1    = (const float*)d_in[10];
  const float* w2    = (const float*)d_in[11];
  const float* b2    = (const float*)d_in[12];
  float* out = (float*)d_out;

  // Workspace carving (256B aligned slices)
  size_t off = 0;
  auto carve = [&](size_t bytes) -> void* {
    void* p = (char*)d_ws + off;
    off += (bytes + 255) & ~(size_t)255;
    return p;
  };
  bf16_t* x_bf    = (bf16_t*)carve((size_t)B_ * T_ * IN_ * 2);   // 16.8 MB
  bf16_t* h0_all  = (bf16_t*)carve((size_t)B_ * T_ * H_  * 2);   // 134 MB
  bf16_t* w_ih0b  = (bf16_t*)carve((size_t)G4_ * IN_ * 2);
  bf16_t* w_hh0b  = (bf16_t*)carve((size_t)G4_ * H_  * 2);
  bf16_t* w_ih1b  = (bf16_t*)carve((size_t)G4_ * H_  * 2);
  bf16_t* w_hh1b  = (bf16_t*)carve((size_t)G4_ * H_  * 2);
  bf16_t* w1b     = (bf16_t*)carve((size_t)H2_ * H_  * 2);
  bf16_t* w2b     = (bf16_t*)carve((size_t)NC_ * H2_ * 2);
  float*  bias0   = (float*) carve((size_t)G4_ * 4);
  float*  bias1   = (float*) carve((size_t)G4_ * 4);
  float*  c0      = (float*) carve((size_t)B_ * H_ * 4);
  float*  c1      = (float*) carve((size_t)B_ * H_ * 4);
  bf16_t* h1buf0  = (bf16_t*)carve((size_t)B_ * H_ * 2);
  bf16_t* h1buf1  = (bf16_t*)carve((size_t)B_ * H_ * 2);
  bf16_t* hmid    = (bf16_t*)carve((size_t)B_ * H2_ * 2);
  bf16_t* h1b[2]  = { h1buf0, h1buf1 };

  // --- one-time (per call) precompute: conversions, bias combine, c init ---
  f2bf_kernel<<<2048, 256, 0, stream>>>(x,     x_bf,   (long long)B_ * T_ * IN_);
  f2bf_kernel<<<256,  256, 0, stream>>>(w_ih0, w_ih0b, (long long)G4_ * IN_);
  f2bf_kernel<<<2048, 256, 0, stream>>>(w_hh0, w_hh0b, (long long)G4_ * H_);
  f2bf_kernel<<<2048, 256, 0, stream>>>(w_ih1, w_ih1b, (long long)G4_ * H_);
  f2bf_kernel<<<2048, 256, 0, stream>>>(w_hh1, w_hh1b, (long long)G4_ * H_);
  f2bf_kernel<<<256,  256, 0, stream>>>(w1,    w1b,    (long long)H2_ * H_);
  f2bf_kernel<<<64,   256, 0, stream>>>(w2,    w2b,    (long long)NC_ * H2_);
  addvec_kernel<<<8,  256, 0, stream>>>(b_ih0, b_hh0, bias0, (long long)G4_);
  addvec_kernel<<<8,  256, 0, stream>>>(b_ih1, b_hh1, bias1, (long long)G4_);
  zerof_kernel<<<512, 256, 0, stream>>>(c0, (long long)B_ * H_);
  zerof_kernel<<<512, 256, 0, stream>>>(c1, (long long)B_ * H_);

  const dim3 step_grid(16, 4);

  // --- layer 0: h state lives inside h0_all (slice t written, t-1 read) ---
  for (int t = 0; t < T_; ++t) {
    const bf16_t* hprev = (t == 0) ? nullptr : (h0_all + (long long)(t - 1) * H_);
    lstm_step_kernel<<<step_grid, 256, 0, stream>>>(
        x_bf + (long long)t * IN_, (long long)T_ * IN_, w_ih0b, IN_,
        hprev, (long long)T_ * H_, w_hh0b, bias0, c0,
        h0_all + (long long)t * H_, (long long)T_ * H_);
  }

  // --- layer 1: consumes h0_all slices; ping-pong h state ---
  for (int t = 0; t < T_; ++t) {
    const bf16_t* hprev = (t == 0) ? nullptr : h1b[(t - 1) & 1];
    lstm_step_kernel<<<step_grid, 256, 0, stream>>>(
        h0_all + (long long)t * H_, (long long)T_ * H_, w_ih1b, H_,
        hprev, (long long)H_, w_hh1b, bias1, c1,
        h1b[t & 1], (long long)H_);
  }

  // --- head: feat = h1 at t = T-1; hmid = relu(feat@w1^T+b1); logits ---
  const bf16_t* feat = h1b[(T_ - 1) & 1];
  // GEMM1: M=256, N=256, K=512 -> 256 tiles / 8 waves = 32 blocks
  head_gemm_kernel<<<32, 256, 0, stream>>>(feat, H_, w1b, b1,
                                           H2_ / 16, H_,
                                           nullptr, hmid, H2_, 1,
                                           (B_ / 16) * (H2_ / 16));
  // GEMM2: M=256, N=64, K=256 -> 64 tiles / 8 waves = 8 blocks
  head_gemm_kernel<<<8, 256, 0, stream>>>(hmid, H2_, w2b, b2,
                                          NC_ / 16, H2_,
                                          out, nullptr, NC_, 0,
                                          (B_ / 16) * (NC_ / 16));
}
